// Gemma3Attention_34943853920517
// MI455X (gfx1250) — matrix operand order repaired and checked by reading, not compiled, not run on hardware
//
#include <hip/hip_runtime.h>
#include <hip/hip_bf16.h>

// ---------------------------------------------------------------------------
// Gemma3 sliding-window attention for MI455X (gfx1250, wave32, WMMA).
// Heavy math via v_wmma_f32_16x16x32_bf16 (fp32 accumulate); all global->LDS
// tile staging via CDNA5 async copies (global_load_async_to_lds_b128).
// ---------------------------------------------------------------------------

typedef unsigned short u16;
typedef unsigned int   u32;
typedef unsigned long long u64;
typedef __attribute__((ext_vector_type(16))) __bf16 bf16x16;
typedef __attribute__((ext_vector_type(8)))  float  f32x8;

#define SEQ    4096
#define DMODEL 2048
#define NH     8
#define NKV    4
#define HD     256
#define WIN    1024
#define NEG_INF (-__builtin_inff())

union Frag {            // 16 bf16 per lane = one WMMA A/B fragment
  uint4   q[2];
  bf16x16 v;
};

static __device__ __forceinline__ f32x8 wmma_bf16(const Frag& a, const Frag& b, f32x8 c) {
  return __builtin_amdgcn_wmma_f32_16x16x32_bf16(false, a.v, false, b.v, (short)0, c, false, false);
}

static __device__ __forceinline__ u16 f2bf(float f) {   // RNE f32 -> bf16
  u32 u = __float_as_uint(f);
  u32 r = u + 0x7FFFu + ((u >> 16) & 1u);
  return (u16)(r >> 16);
}

// CDNA5 async global->LDS copy, 16B per lane. Tracked by ASYNCcnt.
// vdst = LDS byte offset (low 32 bits of generic pointer), vaddr = 64-bit VA.
static __device__ __forceinline__ void async_copy_b128(void* lds, const void* gptr) {
  asm volatile("global_load_async_to_lds_b128 %0, %1, off"
               :: "v"((u32)(size_t)lds), "v"((u64)(size_t)gptr)
               : "memory");
}
static __device__ __forceinline__ void wait_async0() {
  asm volatile("s_wait_asynccnt 0" ::: "memory");
}

static __device__ __forceinline__ float half_max(float v) {  // reduce within 16-lane halves
  #pragma unroll
  for (int m = 1; m < 16; m <<= 1) v = fmaxf(v, __shfl_xor(v, m, 32));
  return v;
}
static __device__ __forceinline__ float half_sum(float v) {
  #pragma unroll
  for (int m = 1; m < 16; m <<= 1) v += __shfl_xor(v, m, 32);
  return v;
}

// ---------------------------------------------------------------------------
// fp32 -> bf16 conversion
// ---------------------------------------------------------------------------
__global__ __launch_bounds__(256) void cvt_bf16_kernel(const float* __restrict__ in,
                                                       u16* __restrict__ out, int n) {
  int i = blockIdx.x * 256 + threadIdx.x;
  if (i < n) out[i] = f2bf(in[i]);
}

// ---------------------------------------------------------------------------
// C[M][N](f32) = A_bf16[M][K] @ B_bf16[N][K]^T    (x @ W^T style)
// BM=128 BN=64 BK=64, 256 threads = 8 waves, each wave a 32x32 tile.
// Double-buffered LDS, async-copy staged; copies overlap WMMA compute.
// ---------------------------------------------------------------------------
#define BM 128
#define BN 64
#define BK 64

__global__ __launch_bounds__(256) void gemm_bf16_kernel(const u16* __restrict__ A,
                                                        const u16* __restrict__ B,
                                                        float* __restrict__ C,
                                                        int M, int N, int K) {
  __shared__ u16 As[2][BM][BK + 8];
  __shared__ u16 Bs[2][BN][BK + 8];
  const int tid = threadIdx.x, lane = tid & 31, wave = tid >> 5;
  const int wx = wave & 3, wy = wave >> 2;
  const int bm0 = blockIdx.x * BM, bn0 = blockIdx.y * BN;
  const int nlocal = lane & 15;
  f32x8 acc[2][2] = {};

  // per-thread staging coordinates (fixed across tiles)
  const int ar[4] = { (tid + 0) >> 3, (tid + 256) >> 3, (tid + 512) >> 3, (tid + 768) >> 3 };
  const int ac4   = (tid & 7) << 3;
  const int br[2] = { (tid + 0) >> 3, (tid + 256) >> 3 };

  auto stage = [&](int k0, int buf) {
    #pragma unroll
    for (int i = 0; i < 4; ++i)                          // A tile: 128x64 bf16
      async_copy_b128(&As[buf][ar[i]][ac4],
                      &A[(size_t)(bm0 + ar[i]) * K + k0 + ac4]);
    #pragma unroll
    for (int i = 0; i < 2; ++i)                          // B tile: 64x64 bf16
      async_copy_b128(&Bs[buf][br[i]][ac4],
                      &B[(size_t)(bn0 + br[i]) * K + k0 + ac4]);
  };

  stage(0, 0);
  int buf = 0;
  for (int k0 = 0; k0 < K; k0 += BK) {
    wait_async0();
    __syncthreads();
    if (k0 + BK < K) stage(k0 + BK, buf ^ 1);            // overlap next copy w/ compute

    #pragma unroll
    for (int kk = 0; kk < BK; kk += 32) {
      Frag a[2], b[2];
      int ac = kk + ((lane < 16) ? 0 : 16);              // A frag: 16 contiguous K
      #pragma unroll
      for (int i = 0; i < 2; ++i) {
        const u16* p = &As[buf][wx * 32 + i * 16 + nlocal][ac];
        a[i].q[0] = *(const uint4*)p;
        a[i].q[1] = *(const uint4*)(p + 8);
      }
      int bc = kk + ((lane < 16) ? 0 : 16);              // B frag: 16 contiguous K
      #pragma unroll
      for (int j = 0; j < 2; ++j) {
        const u16* p = &Bs[buf][wy * 32 + j * 16 + nlocal][bc];
        b[j].q[0] = *(const uint4*)p;
        b[j].q[1] = *(const uint4*)(p + 8);
      }
      #pragma unroll
      for (int i = 0; i < 2; ++i)
        #pragma unroll
        for (int j = 0; j < 2; ++j)
          acc[i][j] = wmma_bf16(a[i], b[j], acc[i][j]);
    }
    buf ^= 1;
  }

  const int radd = (lane >> 4) << 3;                     // C layout: M = r + 8*(lane>=16)
  #pragma unroll
  for (int i = 0; i < 2; ++i)
    #pragma unroll
    for (int j = 0; j < 2; ++j)
      #pragma unroll
      for (int r = 0; r < 8; ++r) {
        int m = bm0 + wx * 32 + i * 16 + r + radd;
        int n = bn0 + wy * 32 + j * 16 + nlocal;
        C[(size_t)m * N + n] = acc[i][j][r];
      }
}

// ---------------------------------------------------------------------------
// Fused RMS-norm + RoPE over one (t, head) row of 256; one wave per row.
// Output bf16 [head][T][256].
// ---------------------------------------------------------------------------
__global__ __launch_bounds__(32) void rmsrope_kernel(const float* __restrict__ X,
                                                     const float* __restrict__ w,
                                                     const int* __restrict__ pos,
                                                     u16* __restrict__ out, int nheads) {
  const int t = blockIdx.x, h = blockIdx.y, lane = threadIdx.x;
  const float* row = X + ((size_t)t * nheads + h) * HD;
  float v[8];
  float4 a = *(const float4*)(row + lane * 8);
  float4 b = *(const float4*)(row + lane * 8 + 4);
  v[0] = a.x; v[1] = a.y; v[2] = a.z; v[3] = a.w;
  v[4] = b.x; v[5] = b.y; v[6] = b.z; v[7] = b.w;

  float ss = 0.f;
  #pragma unroll
  for (int e = 0; e < 8; ++e) ss += v[e] * v[e];
  #pragma unroll
  for (int m = 1; m < 32; m <<= 1) ss += __shfl_xor(ss, m, 32);
  const float sc = rsqrtf(ss * (1.0f / HD) + 1e-6f);
  #pragma unroll
  for (int e = 0; e < 8; ++e) v[e] = v[e] * sc * w[lane * 8 + e];

  const float p = (float)pos[t];
  u16 o[8];
  #pragma unroll
  for (int e = 0; e < 8; e += 2) {                       // interleaved rotate, concat cos/sin (per ref)
    int d0 = lane * 8 + e;
    float th0 = p * __powf(10000.f, -(float)(d0 & 127) * (1.f / 128.f));
    float th1 = p * __powf(10000.f, -(float)((d0 + 1) & 127) * (1.f / 128.f));
    float s0, c0, s1, c1;
    __sincosf(th0, &s0, &c0);
    __sincosf(th1, &s1, &c1);
    o[e]     = f2bf(v[e]     * c0 - v[e + 1] * s0);
    o[e + 1] = f2bf(v[e + 1] * c1 + v[e]     * s1);
  }
  uint4 pk;
  pk.x = o[0] | ((u32)o[1] << 16); pk.y = o[2] | ((u32)o[3] << 16);
  pk.z = o[4] | ((u32)o[5] << 16); pk.w = o[6] | ((u32)o[7] << 16);
  *(uint4*)(out + ((size_t)h * SEQ + t) * HD + lane * 8) = pk;
}

// ---------------------------------------------------------------------------
// V: f32 [T][NKV*256] -> bf16 dim-major [h][d][T] so attention B-frags are
// contiguous-K in LDS.
// ---------------------------------------------------------------------------
__global__ __launch_bounds__(256) void vtr_kernel(const float* __restrict__ V,
                                                  u16* __restrict__ Vt) {
  int i = blockIdx.x * 256 + threadIdx.x;
  if (i >= SEQ * (NKV * HD)) return;
  int t = i >> 10, c = i & 1023;
  int h = c >> 8, d = c & 255;
  Vt[((size_t)(h * HD + d)) * SEQ + t] = f2bf(V[i]);
}

// ---------------------------------------------------------------------------
// Sliding-window flash attention. Block = 256 thr = 8 waves; each wave owns a
// 16-query tile; 32-key chunks async-staged in LDS (K row-major, V dim-major).
// ---------------------------------------------------------------------------
__global__ __launch_bounds__(256) void attn_kernel(const u16* __restrict__ Qr,
                                                   const u16* __restrict__ Kr,
                                                   const u16* __restrict__ Vt,
                                                   u16* __restrict__ Ob) {
  __shared__ u16 Ks[32][HD + 8];          // [key][d]
  __shared__ u16 Vs[HD][32 + 8];          // [d][key]
  __shared__ u16 Ps[8][16][32 + 8];       // per-wave P re-layout scratch

  const int tid = threadIdx.x, lane = tid & 31, wave = tid >> 5;
  const int head = blockIdx.y, kvh = head >> 1;
  const int q0 = blockIdx.x * 128;
  const int qw = q0 + wave * 16;
  const int nlocal = lane & 15;
  const int radd = (lane >> 4) << 3;
  const float SCALE = 0.0625f;            // 1/sqrt(256)

  const u16* Qh = Qr + (size_t)head * SEQ * HD;
  const u16* Kh = Kr + (size_t)kvh  * SEQ * HD;
  const u16* Vh = Vt + (size_t)kvh  * HD  * SEQ;

  Frag qf[8];                             // 16 queries x 256 dims, preloaded
  {
    const u16* qrow = Qh + (size_t)(qw + nlocal) * HD + ((lane < 16) ? 0 : 16);
    #pragma unroll
    for (int d = 0; d < 8; ++d) {
      qf[d].q[0] = *(const uint4*)(qrow + d * 32);
      qf[d].q[1] = *(const uint4*)(qrow + d * 32 + 8);
    }
  }

  f32x8 oacc[16] = {};
  float mrow[8], lrow[8];
  #pragma unroll
  for (int r = 0; r < 8; ++r) { mrow[r] = NEG_INF; lrow[r] = 0.f; }

  int lo = q0 - (WIN - 1); if (lo < 0) lo = 0;
  const int cstart = lo >> 5;
  const int cend   = (q0 + 127) >> 5;

  for (int ck = cstart; ck <= cend; ++ck) {
    const int kb = ck << 5;
    __syncthreads();                                     // all waves done reading prev chunk
    #pragma unroll
    for (int i = 0; i < 4; ++i) {                        // 32 keys x 256 dims
      int idx = tid + i * 256;
      int r = idx >> 5, c = (idx & 31) << 3;
      async_copy_b128(&Ks[r][c], &Kh[(size_t)(kb + r) * HD + c]);
    }
    #pragma unroll
    for (int i = 0; i < 4; ++i) {                        // 256 dims x 32 keys
      int idx = tid + i * 256;
      int r = idx >> 2, c = (idx & 3) << 3;
      async_copy_b128(&Vs[r][c], &Vh[(size_t)r * SEQ + kb + c]);
    }
    wait_async0();
    __syncthreads();

    // S = Q K^T : two 16x16 score tiles over the 32-key chunk
    f32x8 s0 = {}, s1 = {};
    {
      const int bc = (lane < 16) ? 0 : 16;
      #pragma unroll
      for (int d = 0; d < 8; ++d) {
        Frag k0f, k1f;
        const u16* p0 = &Ks[nlocal][d * 32 + bc];
        k0f.q[0] = *(const uint4*)p0; k0f.q[1] = *(const uint4*)(p0 + 8);
        const u16* p1 = &Ks[16 + nlocal][d * 32 + bc];
        k1f.q[0] = *(const uint4*)p1; k1f.q[1] = *(const uint4*)(p1 + 8);
        s0 = wmma_bf16(qf[d], k0f, s0);
        s1 = wmma_bf16(qf[d], k1f, s1);
      }
    }

    // online softmax (per-row state in elements r, replicated over 16-lane halves)
    float alpha[8];
    #pragma unroll
    for (int r = 0; r < 8; ++r) {
      const int qi  = qw + r + radd;
      const int k0i = kb + nlocal;
      const int k1i = kb + 16 + nlocal;
      float v0 = s0[r] * SCALE, v1 = s1[r] * SCALE;
      v0 = (k0i <= qi && k0i >= qi - (WIN - 1)) ? v0 : NEG_INF;
      v1 = (k1i <= qi && k1i >= qi - (WIN - 1)) ? v1 : NEG_INF;
      float rm   = half_max(fmaxf(v0, v1));
      float mold = mrow[r];
      float mnew = fmaxf(mold, rm);
      float a  = (mold == NEG_INF) ? 0.f : __expf(mold - mnew);
      float p0 = (v0   == NEG_INF) ? 0.f : __expf(v0 - mnew);
      float p1 = (v1   == NEG_INF) ? 0.f : __expf(v1 - mnew);
      float rs = half_sum(p0 + p1);
      mrow[r]  = mnew;
      lrow[r]  = lrow[r] * a + rs;
      alpha[r] = a;
      Ps[wave][r + radd][nlocal]      = f2bf(p0);        // C-layout -> row-major P
      Ps[wave][r + radd][16 + nlocal] = f2bf(p1);
    }
    #pragma unroll
    for (int nt = 0; nt < 16; ++nt)
      #pragma unroll
      for (int r = 0; r < 8; ++r)
        oacc[nt][r] *= alpha[r];

    asm volatile("s_wait_dscnt 0" ::: "memory");         // P stores visible to frag reloads

    Frag pf;                                             // P as A-frag (16x32, K=32 keys)
    {
      const u16* p = &Ps[wave][nlocal][(lane < 16) ? 0 : 16];
      pf.q[0] = *(const uint4*)p;
      pf.q[1] = *(const uint4*)(p + 8);
    }
    {
      const int vc = (lane < 16) ? 0 : 16;
      #pragma unroll
      for (int nt = 0; nt < 16; ++nt) {                  // O += P @ V over all 256 dims
        Frag vf;
        const u16* p = &Vs[nt * 16 + nlocal][vc];
        vf.q[0] = *(const uint4*)p;
        vf.q[1] = *(const uint4*)(p + 8);
        oacc[nt] = wmma_bf16(pf, vf, oacc[nt]);
      }
    }
  }

  #pragma unroll
  for (int nt = 0; nt < 16; ++nt)
    #pragma unroll
    for (int r = 0; r < 8; ++r) {
      float o = oacc[nt][r] / lrow[r];
      int m = qw + r + radd;
      Ob[(size_t)m * DMODEL + head * HD + nt * 16 + nlocal] = f2bf(o);
    }
}

// ---------------------------------------------------------------------------
// Host-side orchestration (graph-capture safe: only kernel launches on stream)
// ---------------------------------------------------------------------------
extern "C" void kernel_launch(void* const* d_in, const int* in_sizes, int n_in,
                              void* d_out, int out_size, void* d_ws, size_t ws_size,
                              hipStream_t stream) {
  const float* x   = (const float*)d_in[0];
  const int*   pos = (const int*)d_in[1];
  const float* Wq  = (const float*)d_in[2];
  const float* Wk  = (const float*)d_in[3];
  const float* Wv  = (const float*)d_in[4];
  const float* Wo  = (const float*)d_in[5];
  const float* qnw = (const float*)d_in[6];
  const float* knw = (const float*)d_in[7];
  float* out = (float*)d_out;

  char* ws = (char*)d_ws;
  size_t off = 0;
  auto alloc = [&](size_t bytes) {
    void* p = ws + off;
    off = (off + bytes + 255) & ~(size_t)255;
    return p;
  };
  u16* xb    = (u16*)alloc((size_t)SEQ * DMODEL * 2);        // x bf16
  u16* wqb   = (u16*)alloc((size_t)DMODEL * DMODEL * 2);
  u16* wkb   = (u16*)alloc((size_t)(NKV * HD) * DMODEL * 2);
  u16* wvb   = (u16*)alloc((size_t)(NKV * HD) * DMODEL * 2);
  u16* wob   = (u16*)alloc((size_t)DMODEL * DMODEL * 2);
  float* Qf  = (float*)alloc((size_t)SEQ * DMODEL * 4);
  float* Kf  = (float*)alloc((size_t)SEQ * (NKV * HD) * 4);
  float* Vf  = (float*)alloc((size_t)SEQ * (NKV * HD) * 4);
  u16* Qr    = (u16*)alloc((size_t)NH  * SEQ * HD * 2);      // [h][t][d] bf16
  u16* Kr    = (u16*)alloc((size_t)NKV * SEQ * HD * 2);
  u16* Vtb   = (u16*)alloc((size_t)NKV * HD * SEQ * 2);      // [h][d][t] bf16
  u16* attnb = (u16*)alloc((size_t)SEQ * DMODEL * 2);

  auto cvt = [&](const float* src, u16* dst, int n) {
    cvt_bf16_kernel<<<(n + 255) / 256, 256, 0, stream>>>(src, dst, n);
  };
  cvt(x,  xb,  SEQ * DMODEL);
  cvt(Wq, wqb, DMODEL * DMODEL);
  cvt(Wk, wkb, (NKV * HD) * DMODEL);
  cvt(Wv, wvb, (NKV * HD) * DMODEL);
  cvt(Wo, wob, DMODEL * DMODEL);

  // QKV projections: x @ W^T
  gemm_bf16_kernel<<<dim3(SEQ / BM, DMODEL / BN),   256, 0, stream>>>(xb, wqb, Qf, SEQ, DMODEL,   DMODEL);
  gemm_bf16_kernel<<<dim3(SEQ / BM, (NKV*HD) / BN), 256, 0, stream>>>(xb, wkb, Kf, SEQ, NKV * HD, DMODEL);
  gemm_bf16_kernel<<<dim3(SEQ / BM, (NKV*HD) / BN), 256, 0, stream>>>(xb, wvb, Vf, SEQ, NKV * HD, DMODEL);

  // RMS-norm + RoPE -> bf16, and V transpose -> bf16 [h][d][t]
  rmsrope_kernel<<<dim3(SEQ, NH),  32, 0, stream>>>(Qf, qnw, pos, Qr, NH);
  rmsrope_kernel<<<dim3(SEQ, NKV), 32, 0, stream>>>(Kf, knw, pos, Kr, NKV);
  vtr_kernel<<<(SEQ * NKV * HD + 255) / 256, 256, 0, stream>>>(Vf, Vtb);

  // Sliding-window flash attention
  attn_kernel<<<dim3(SEQ / 128, NH), 256, 0, stream>>>(Qr, Kr, Vtb, attnb);

  // Output projection: attn @ Wo^T -> f32 d_out
  gemm_bf16_kernel<<<dim3(SEQ / BM, DMODEL / BN), 256, 0, stream>>>(attnb, wob, out, SEQ, DMODEL, DMODEL);
}
